// Transformer_33569464385867
// MI455X (gfx1250) — compile-verified, matrix operand order checked by reading, not run on hardware
//
#include <hip/hip_runtime.h>

typedef __bf16 bf16_t;
typedef __attribute__((ext_vector_type(16))) __bf16 v16bf;
typedef __attribute__((ext_vector_type(8)))  __bf16 v8bf;
typedef __attribute__((ext_vector_type(8)))  float  v8f;
#define LL long long

// ---- model dims (match reference) ----
constexpr int Bb = 4, Ss = 512, Dd = 2048;
constexpr int Hh = 16, KVv = 4, HDd = 128, REP = Hh / KVv;
constexpr int FFNf = 5632, Ee = 8, Rr = 16;
constexpr int Tt = Bb * Ss;
constexpr float LSCALE = 32.0f / 16.0f;   // 2.0
constexpr float EPSc   = 1e-5f;

// ---- GEMM tiling ----
constexpr int BM = 128, BN = 128, BK = 64, PADB = 8;  // row stride 72 elems = 144 B (16B aligned)

#if defined(__HIP_DEVICE_COMPILE__) && __has_builtin(__builtin_amdgcn_global_load_async_to_lds_b128)
#define HAVE_ASYNC_G2L 1
#else
#define HAVE_ASYNC_G2L 0
#endif

__device__ __forceinline__ void wait_async0() {
#if defined(__HIP_DEVICE_COMPILE__)
#if __has_builtin(__builtin_amdgcn_s_wait_asynccnt)
  __builtin_amdgcn_s_wait_asynccnt(0);
#else
  asm volatile("s_wait_asynccnt 0" ::: "memory");
#endif
#endif
}

#if HAVE_ASYNC_G2L
typedef int v4i_t __attribute__((vector_size(16)));
typedef __attribute__((address_space(1))) v4i_t* g_v4i_p;
typedef __attribute__((address_space(3))) v4i_t* l_v4i_p;
__device__ __forceinline__ void async_g2l_b128(const void* g, void* l) {
  __builtin_amdgcn_global_load_async_to_lds_b128((g_v4i_p)(g), (l_v4i_p)(l), 0, 0);
}
#endif

__device__ __forceinline__ LL boff(int z, int qdiv, LL qs, int rmod, LL rs) {
  return (LL)(z / qdiv) * qs + (LL)(z % rmod) * rs;
}

union FragAB { v16bf v; v8bf h[2]; };

// C[M,N] = alpha * A[M,K] @ B[N,K]^T (+ beta * Cin),  bf16 in, f32 accumulate.
// Double-buffered LDS; interior tiles stream via global_load_async_to_lds_b128.
// Requires K % 8 == 0 (true for every GEMM in this model).
// Batched via blockIdx.z with (z/div)*stride (+ (z%mod)*stride for C) offsets.
__global__ __launch_bounds__(256)
void gemm_bf16_nt(const bf16_t* __restrict__ A, int lda, int adiv, LL aqs,
                  const bf16_t* __restrict__ Bm, int ldb, int bdiv, LL bqs,
                  const float* __restrict__ Cin, float* __restrict__ C, int ldc,
                  int cdiv, LL cqs, int cmod, LL cms,
                  bf16_t* __restrict__ Cb, int ldcb,
                  int M, int N, int K, float alpha, float beta)
{
  __shared__ bf16_t sA[2][BM][BK + PADB];
  __shared__ bf16_t sB[2][BN][BK + PADB];

  const int z = blockIdx.z;
  A  += boff(z, adiv, aqs, 1, 0);
  Bm += boff(z, bdiv, bqs, 1, 0);
  const LL coff = boff(z, cdiv, cqs, cmod, cms);
  C += coff;
  if (Cin) Cin += coff;
  if (Cb)  Cb  += coff;

  const int tid  = threadIdx.x;
  const int lane = tid & 31;
  const int wave = tid >> 5;
  const int wm   = (wave & 1) * 64;   // wave M base inside block tile (2 waves)
  const int wn   = (wave >> 1) * 32;  // wave N base inside block tile (4 waves)

  v8f acc[4][2];
#pragma unroll
  for (int i = 0; i < 4; ++i)
#pragma unroll
    for (int j = 0; j < 2; ++j)
#pragma unroll
      for (int k = 0; k < 8; ++k) acc[i][j][k] = 0.0f;

  // staging: 256 threads, 128 rows x 64 cols per tile -> 2 threads/row, 32 elems each
  const int lr = tid >> 1;
  const int k0 = (tid & 1) * 32;
  const LL  gA = (LL)blockIdx.y * BM + lr;
  const LL  gB = (LL)blockIdx.x * BN + lr;

  // fully in-bounds block tile + K divisible -> unguarded (async) staging
  const bool interior = ((LL)blockIdx.y * BM + BM <= (LL)M) &&
                        ((LL)blockIdx.x * BN + BN <= (LL)N) &&
                        (K % BK == 0);

  // WMMA fragment addressing (16-bit A/B 16x32 layout, wave32)
  const int fr = lane & 15;
  const int kg = (lane >> 4) << 3;    // 0 or 8

  auto stage_sync = [&](int buf, int kt) {
    __builtin_prefetch(A + (gA < M ? gA : 0) * lda + kt, 0, 1);
#pragma unroll
    for (int j = 0; j < 4; ++j) {
      const int kk = kt + k0 + j * 8;
      uint4 va = make_uint4(0u, 0u, 0u, 0u);
      if (gA < M && kk + 8 <= K)
        va = *reinterpret_cast<const uint4*>(A + gA * lda + kk);
      *reinterpret_cast<uint4*>(&sA[buf][lr][k0 + j * 8]) = va;
      uint4 vb = make_uint4(0u, 0u, 0u, 0u);
      if (gB < N && kk + 8 <= K)
        vb = *reinterpret_cast<const uint4*>(Bm + gB * ldb + kk);
      *reinterpret_cast<uint4*>(&sB[buf][lr][k0 + j * 8]) = vb;
    }
  };

  auto stage = [&](int buf, int kt) {
#if HAVE_ASYNC_G2L
    if (interior) {
      const bf16_t* ga = A  + gA * lda + kt + k0;
      const bf16_t* gb = Bm + gB * ldb + kt + k0;
      bf16_t* la = &sA[buf][lr][k0];
      bf16_t* lb = &sB[buf][lr][k0];
#pragma unroll
      for (int j = 0; j < 4; ++j) {
        async_g2l_b128(ga + j * 8, la + j * 8);
        async_g2l_b128(gb + j * 8, lb + j * 8);
      }
      return;
    }
#endif
    stage_sync(buf, kt);
  };
  auto stage_wait = [&]() {
#if HAVE_ASYNC_G2L
    if (interior) wait_async0();
#endif
  };

  stage(0, 0);
  stage_wait();
  __syncthreads();

  int buf = 0;
  for (int kt = 0; kt < K; kt += BK) {
    const bool hasNext = (kt + BK) < K;
    if (hasNext) stage(buf ^ 1, kt + BK);

#pragma unroll
    for (int ks = 0; ks < 2; ++ks) {
      const int ko = ks * 32;
      FragAB fa[4], fb[2];
#pragma unroll
      for (int mi = 0; mi < 4; ++mi) {
        const bf16_t* p = &sA[buf][wm + mi * 16 + fr][ko];
        fa[mi].h[0] = *reinterpret_cast<const v8bf*>(p + kg);
        fa[mi].h[1] = *reinterpret_cast<const v8bf*>(p + 16 + kg);
      }
#pragma unroll
      for (int ni = 0; ni < 2; ++ni) {
        const bf16_t* p = &sB[buf][wn + ni * 16 + fr][ko];
        fb[ni].h[0] = *reinterpret_cast<const v8bf*>(p + kg);
        fb[ni].h[1] = *reinterpret_cast<const v8bf*>(p + 16 + kg);
      }
#pragma unroll
      for (int mi = 0; mi < 4; ++mi)
#pragma unroll
        for (int ni = 0; ni < 2; ++ni)
          acc[mi][ni] = __builtin_amdgcn_wmma_f32_16x16x32_bf16(
              false, fa[mi].v, false, fb[ni].v, (short)0, acc[mi][ni], false, false);
    }

    if (hasNext) {
      stage_wait();
      __syncthreads();
    }
    buf ^= 1;
  }

  // epilogue: C layout lane = (m>=8)*16 + n, vgpr v -> row m = 8*(lane>>4)+v
  const int en  = lane & 15;
  const int em0 = (lane >> 4) << 3;
#pragma unroll
  for (int mi = 0; mi < 4; ++mi) {
#pragma unroll
    for (int ni = 0; ni < 2; ++ni) {
      const LL gn = (LL)blockIdx.x * BN + wn + ni * 16 + en;
      if (gn >= N) continue;
#pragma unroll
      for (int v = 0; v < 8; ++v) {
        const LL gm = (LL)blockIdx.y * BM + wm + mi * 16 + em0 + v;
        if (gm >= M) continue;
        float r = alpha * acc[mi][ni][v];
        if (Cin) r += beta * Cin[gm * (LL)ldc + gn];
        C[gm * (LL)ldc + gn] = r;
        if (Cb) Cb[gm * (LL)ldcb + gn] = (bf16_t)r;
      }
    }
  }
}

// ---- elementwise / reduction kernels ----
__global__ void cast_f32_bf16(const float* __restrict__ x, bf16_t* __restrict__ y, LL n) {
  LL i = (LL)blockIdx.x * blockDim.x + threadIdx.x;
  LL stride = (LL)gridDim.x * blockDim.x;
  for (; i < n; i += stride) y[i] = (bf16_t)x[i];
}

// normOut = rmsnorm(x + res) * w ; sumOut (optional) = x + res
__global__ __launch_bounds__(256)
void rmsnorm_k(const float* __restrict__ x, const float* __restrict__ res,
               const float* __restrict__ w, float* __restrict__ sumOut,
               bf16_t* __restrict__ normOut, int ncols) {
  __shared__ float red[256];
  const LL row = blockIdx.x;
  const float* xr = x + row * ncols;
  const float* rr = res ? res + row * ncols : nullptr;
  float ss = 0.f;
  for (int i = threadIdx.x; i < ncols; i += 256) {
    float v = xr[i] + (rr ? rr[i] : 0.f);
    ss += v * v;
  }
  red[threadIdx.x] = ss; __syncthreads();
  for (int s = 128; s > 0; s >>= 1) {
    if (threadIdx.x < s) red[threadIdx.x] += red[threadIdx.x + s];
    __syncthreads();
  }
  const float rstd = rsqrtf(red[0] / (float)ncols + EPSc);
  for (int i = threadIdx.x; i < ncols; i += 256) {
    float v = xr[i] + (rr ? rr[i] : 0.f);
    if (sumOut) sumOut[row * ncols + i] = v;
    normOut[row * ncols + i] = (bf16_t)(v * rstd * w[i]);
  }
}

// RoPE: x (B,S,nh,HD) f32 -> o (B,nh,S,HD) bf16, interleaved even/odd pairs
__global__ void rope_k(const float* __restrict__ x, const float* __restrict__ c,
                       const float* __restrict__ s, bf16_t* __restrict__ o, int nh) {
  const LL total = (LL)Bb * Ss * nh * (HDd / 2);
  LL idx = (LL)blockIdx.x * blockDim.x + threadIdx.x;
  if (idx >= total) return;
  int i  = (int)(idx % (HDd / 2));
  int h  = (int)((idx / (HDd / 2)) % nh);
  int sq = (int)((idx / ((LL)(HDd / 2) * nh)) % Ss);
  int b  = (int)(idx / ((LL)(HDd / 2) * nh * Ss));
  LL inBase = (((LL)(b * Ss + sq) * nh) + h) * HDd + 2 * i;
  float x1 = x[inBase], x2 = x[inBase + 1];
  float co = c[sq * (HDd / 2) + i], si = s[sq * (HDd / 2) + i];
  LL oBase = (((LL)(b * nh + h) * Ss) + sq) * HDd + 2 * i;
  o[oBase]     = (bf16_t)(x1 * co - x2 * si);
  o[oBase + 1] = (bf16_t)(x1 * si + x2 * co);
}

// xv (B,S,KV,HD) f32 -> vt (B,KV,HD,S) bf16  (transposed so P@V uses the NT GEMM)
__global__ void vtrans_k(const float* __restrict__ xv, bf16_t* __restrict__ vt) {
  const LL total = (LL)Bb * KVv * HDd * Ss;
  LL idx = (LL)blockIdx.x * blockDim.x + threadIdx.x;
  if (idx >= total) return;
  int sq = (int)(idx % Ss);
  int d  = (int)((idx / Ss) % HDd);
  int kv = (int)((idx / ((LL)Ss * HDd)) % KVv);
  int b  = (int)(idx / ((LL)Ss * HDd * KVv));
  vt[idx] = (bf16_t)xv[(((LL)(b * Ss + sq) * KVv) + kv) * HDd + d];
}

// rowwise softmax over S=512 with additive mask; one block per row, 2 elems/thread
__global__ __launch_bounds__(256)
void softmax_k(const float* __restrict__ scores, const float* __restrict__ mask,
               bf16_t* __restrict__ attn) {
  __shared__ float red[256];
  const LL row = blockIdx.x;            // (b*H + h)*S + s
  const int sq = (int)(row % Ss);
  const float* r = scores + row * Ss;
  const float* m = mask + (LL)sq * Ss;
  float v0 = r[threadIdx.x]       + m[threadIdx.x];
  float v1 = r[threadIdx.x + 256] + m[threadIdx.x + 256];
  red[threadIdx.x] = fmaxf(v0, v1); __syncthreads();
  for (int s2 = 128; s2 > 0; s2 >>= 1) {
    if (threadIdx.x < s2) red[threadIdx.x] = fmaxf(red[threadIdx.x], red[threadIdx.x + s2]);
    __syncthreads();
  }
  const float mx = red[0]; __syncthreads();
  float e0 = __expf(v0 - mx), e1 = __expf(v1 - mx);
  red[threadIdx.x] = e0 + e1; __syncthreads();
  for (int s2 = 128; s2 > 0; s2 >>= 1) {
    if (threadIdx.x < s2) red[threadIdx.x] += red[threadIdx.x + s2];
    __syncthreads();
  }
  const float inv = 1.f / red[0];
  attn[row * Ss + threadIdx.x]       = (bf16_t)(e0 * inv);
  attn[row * Ss + threadIdx.x + 256] = (bf16_t)(e1 * inv);
}

// top-2 routing: route[t,e] = normalized softmax prob at top-2 indices, else 0
__global__ void router_k(const float* __restrict__ logits, float* __restrict__ route) {
  int t = blockIdx.x * blockDim.x + threadIdx.x;
  if (t >= Tt) return;
  const float* l = logits + (LL)t * Ee;
  int i0 = 0, i1 = 0; float b0 = -1e30f, b1 = -1e30f;
  for (int e = 0; e < Ee; ++e) {
    float v = l[e];
    if (v > b0)      { b1 = b0; i1 = i0; b0 = v; i0 = e; }
    else if (v > b1) { b1 = v;  i1 = e; }
  }
  float e1 = __expf(b1 - b0);
  float sum = 1.0f + e1;
  for (int e = 0; e < Ee; ++e) route[(LL)t * Ee + e] = 0.f;
  route[(LL)t * Ee + i0] = 1.0f / sum;
  route[(LL)t * Ee + i1] = e1 / sum;
}

__global__ void silu_mul_k(const float* __restrict__ g1, const float* __restrict__ g3,
                           bf16_t* __restrict__ gh, LL n) {
  LL i = (LL)blockIdx.x * blockDim.x + threadIdx.x;
  LL stride = (LL)gridDim.x * blockDim.x;
  for (; i < n; i += stride) {
    float a = g1[i];
    float sg = a / (1.f + __expf(-a));
    gh[i] = (bf16_t)(sg * g3[i]);
  }
}

// acc = beta*acc + route[m,e] * oe   (per-token routed accumulate)
__global__ void scale_acc_k(float* __restrict__ acc, const float* __restrict__ oe,
                            const float* __restrict__ route, int e, float beta, LL n) {
  LL i = (LL)blockIdx.x * blockDim.x + threadIdx.x;
  LL stride = (LL)gridDim.x * blockDim.x;
  for (; i < n; i += stride) {
    LL mrow = i / Dd;
    float prev = (beta != 0.f) ? acc[i] : 0.f;
    acc[i] = prev + route[mrow * Ee + e] * oe[i];
  }
}

__global__ void add2_k(float* __restrict__ out, const float* __restrict__ a,
                       const float* __restrict__ b, LL n) {
  LL i = (LL)blockIdx.x * blockDim.x + threadIdx.x;
  LL stride = (LL)gridDim.x * blockDim.x;
  for (; i < n; i += stride) out[i] = a[i] + b[i];
}

// ------------------------------------------------------------------
extern "C" void kernel_launch(void* const* d_in, const int* in_sizes, int n_in,
                              void* d_out, int out_size, void* d_ws, size_t ws_size,
                              hipStream_t stream) {
  (void)in_sizes; (void)n_in; (void)out_size; (void)ws_size;

  const float* f_data = (const float*)d_in[0];
  const float* f_mask = (const float*)d_in[1];
  const float* f_cos  = (const float*)d_in[2];
  const float* f_sin  = (const float*)d_in[3];
  const float* f_attw = (const float*)d_in[4];
  const float* f_ffnw = (const float*)d_in[5];
  const float* f_wq   = (const float*)d_in[6];
  const float* f_wk   = (const float*)d_in[7];
  const float* f_wv   = (const float*)d_in[8];
  const float* f_wo   = (const float*)d_in[9];
  const float* f_lqA  = (const float*)d_in[10];
  const float* f_lqB  = (const float*)d_in[11];
  const float* f_lkA  = (const float*)d_in[12];
  const float* f_lkB  = (const float*)d_in[13];
  const float* f_lvA  = (const float*)d_in[14];
  const float* f_lvB  = (const float*)d_in[15];
  const float* f_loA  = (const float*)d_in[16];
  const float* f_loB  = (const float*)d_in[17];
  const float* f_w1   = (const float*)d_in[18];
  const float* f_w2   = (const float*)d_in[19];
  const float* f_w3   = (const float*)d_in[20];
  const float* f_gate = (const float*)d_in[21];
  const float* f_eA1  = (const float*)d_in[22];
  const float* f_eB1  = (const float*)d_in[23];
  const float* f_eA2  = (const float*)d_in[24];
  const float* f_eB2  = (const float*)d_in[25];
  const float* f_eA3  = (const float*)d_in[26];
  const float* f_eB3  = (const float*)d_in[27];

  // ---- workspace sub-allocator ----
  char* wsbase = (char*)d_ws;
  size_t wsoff = 0;
  auto alloc = [&](size_t bytes) -> void* {
    wsoff = (wsoff + 255) & ~(size_t)255;
    void* p = wsbase + wsoff;
    wsoff += bytes;
    return p;
  };
  auto elgrid = [](LL n) { return dim3((unsigned)((n + 255) / 256)); };
  auto tob = [&](const float* src, LL n) -> bf16_t* {
    bf16_t* d = (bf16_t*)alloc(n * sizeof(bf16_t));
    cast_f32_bf16<<<elgrid(n), 256, 0, stream>>>(src, d, n);
    return d;
  };

  auto gemm = [&](const bf16_t* A, int lda, const bf16_t* Bm, int ldb,
                  float* C, int ldc, int M, int N, int K,
                  float alpha, const float* Cin, float beta,
                  bf16_t* Cb, int ldcb) {
    dim3 g((N + BN - 1) / BN, (M + BM - 1) / BM, 1);
    gemm_bf16_nt<<<g, 256, 0, stream>>>(A, lda, 1, 0, Bm, ldb, 1, 0,
                                        Cin, C, ldc, 1, 0, 1, 0,
                                        Cb, ldcb, M, N, K, alpha, beta);
  };
  auto gemmB = [&](const bf16_t* A, int lda, int adiv, LL aqs,
                   const bf16_t* Bm, int ldb, int bdiv, LL bqs,
                   float* C, int ldc, int cdiv, LL cqs, int cmod, LL cms,
                   bf16_t* Cb, int ldcb,
                   int M, int N, int K, float alpha, int batch) {
    dim3 g((N + BN - 1) / BN, (M + BM - 1) / BM, batch);
    gemm_bf16_nt<<<g, 256, 0, stream>>>(A, lda, adiv, aqs, Bm, ldb, bdiv, bqs,
                                        nullptr, C, ldc, cdiv, cqs, cmod, cms,
                                        Cb, ldcb, M, N, K, alpha, 0.f);
  };

  // ---- weights -> bf16 (ws) ----
  bf16_t* wq_b  = tob(f_wq,  (LL)Dd * Dd);
  bf16_t* wk_b  = tob(f_wk,  (LL)KVv * HDd * Dd);
  bf16_t* wv_b  = tob(f_wv,  (LL)KVv * HDd * Dd);
  bf16_t* wo_b  = tob(f_wo,  (LL)Dd * Dd);
  bf16_t* lqA_b = tob(f_lqA, (LL)Rr * Dd);
  bf16_t* lqB_b = tob(f_lqB, (LL)Dd * Rr);
  bf16_t* lkA_b = tob(f_lkA, (LL)Rr * Dd);
  bf16_t* lkB_b = tob(f_lkB, (LL)KVv * HDd * Rr);
  bf16_t* lvA_b = tob(f_lvA, (LL)Rr * Dd);
  bf16_t* lvB_b = tob(f_lvB, (LL)KVv * HDd * Rr);
  bf16_t* loA_b = tob(f_loA, (LL)Rr * Dd);
  bf16_t* loB_b = tob(f_loB, (LL)Dd * Rr);
  bf16_t* w1_b  = tob(f_w1,  (LL)FFNf * Dd);
  bf16_t* w2_b  = tob(f_w2,  (LL)Dd * FFNf);
  bf16_t* w3_b  = tob(f_w3,  (LL)FFNf * Dd);
  bf16_t* gate_b= tob(f_gate,(LL)Ee * Dd);
  bf16_t* eA1_b = tob(f_eA1, (LL)Ee * Rr * Dd);
  bf16_t* eB1_b = tob(f_eB1, (LL)Ee * FFNf * Rr);
  bf16_t* eA2_b = tob(f_eA2, (LL)Ee * Rr * FFNf);
  bf16_t* eB2_b = tob(f_eB2, (LL)Ee * Dd * Rr);
  bf16_t* eA3_b = tob(f_eA3, (LL)Ee * Rr * Dd);
  bf16_t* eB3_b = tob(f_eB3, (LL)Ee * FFNf * Rr);

  // ---- activation scratch ----
  bf16_t* h_b    = (bf16_t*)alloc((LL)Tt * Dd * 2);
  float*  xq_f   = (float*) alloc((LL)Tt * Dd * 4);
  float*  xk_f   = (float*) alloc((LL)Tt * KVv * HDd * 4);
  float*  xv_f   = (float*) alloc((LL)Tt * KVv * HDd * 4);
  bf16_t* tq_b   = (bf16_t*)alloc((LL)Tt * Rr * 2);
  float*  tscr_f = (float*) alloc((LL)Tt * Rr * 4);
  bf16_t* q_r    = (bf16_t*)alloc((LL)Bb * Hh * Ss * HDd * 2);
  bf16_t* k_r    = (bf16_t*)alloc((LL)Bb * KVv * Ss * HDd * 2);
  bf16_t* v_t    = (bf16_t*)alloc((LL)Bb * KVv * HDd * Ss * 2);
  float*  scr_f  = (float*) alloc((LL)Bb * Hh * Ss * Ss * 4);
  bf16_t* attn_b = (bf16_t*)alloc((LL)Bb * Hh * Ss * Ss * 2);
  float*  ao_f   = (float*) alloc((LL)Tt * Dd * 4);
  bf16_t* ao_b   = (bf16_t*)alloc((LL)Tt * Dd * 2);
  float*  woo_f  = (float*) alloc((LL)Tt * Dd * 4);
  float*  data2  = (float*) alloc((LL)Tt * Dd * 4);
  bf16_t* hn_b   = (bf16_t*)alloc((LL)Tt * Dd * 2);
  float*  lg_f   = (float*) alloc((LL)Tt * Ee * 4);
  float*  rt_f   = (float*) alloc((LL)Tt * Ee * 4);
  float*  h1_f   = (float*) alloc((LL)Tt * FFNf * 4);
  float*  h3_f   = (float*) alloc((LL)Tt * FFNf * 4);
  float*  g1_f   = (float*) alloc((LL)Tt * FFNf * 4);
  float*  g3_f   = (float*) alloc((LL)Tt * FFNf * 4);
  bf16_t* gh_b   = (bf16_t*)alloc((LL)Tt * FFNf * 2);
  bf16_t* t2_b   = (bf16_t*)alloc((LL)Tt * Rr * 2);
  float*  oe_f   = (float*) alloc((LL)Tt * Dd * 4);
  float*  acc_f  = (float*) alloc((LL)Tt * Dd * 4);

  // ---- 1) pre-attention rmsnorm ----
  rmsnorm_k<<<Tt, 256, 0, stream>>>(f_data, nullptr, f_attw, nullptr, h_b, Dd);

  // ---- 2) q/k/v projections + LoRA ----
  gemm(h_b, Dd, wq_b, Dd, xq_f, Dd, Tt, Dd, Dd, 1.f, nullptr, 0.f, nullptr, 0);
  gemm(h_b, Dd, lqA_b, Dd, tscr_f, Rr, Tt, Rr, Dd, 1.f, nullptr, 0.f, tq_b, Rr);
  gemm(tq_b, Rr, lqB_b, Rr, xq_f, Dd, Tt, Dd, Rr, LSCALE, xq_f, 1.f, nullptr, 0);

  gemm(h_b, Dd, wk_b, Dd, xk_f, KVv*HDd, Tt, KVv*HDd, Dd, 1.f, nullptr, 0.f, nullptr, 0);
  gemm(h_b, Dd, lkA_b, Dd, tscr_f, Rr, Tt, Rr, Dd, 1.f, nullptr, 0.f, tq_b, Rr);
  gemm(tq_b, Rr, lkB_b, Rr, xk_f, KVv*HDd, Tt, KVv*HDd, Rr, LSCALE, xk_f, 1.f, nullptr, 0);

  gemm(h_b, Dd, wv_b, Dd, xv_f, KVv*HDd, Tt, KVv*HDd, Dd, 1.f, nullptr, 0.f, nullptr, 0);
  gemm(h_b, Dd, lvA_b, Dd, tscr_f, Rr, Tt, Rr, Dd, 1.f, nullptr, 0.f, tq_b, Rr);
  gemm(tq_b, Rr, lvB_b, Rr, xv_f, KVv*HDd, Tt, KVv*HDd, Rr, LSCALE, xv_f, 1.f, nullptr, 0);

  // ---- 3) RoPE + layout shuffles ----
  rope_k<<<elgrid((LL)Bb*Ss*Hh*(HDd/2)), 256, 0, stream>>>(xq_f, f_cos, f_sin, q_r, Hh);
  rope_k<<<elgrid((LL)Bb*Ss*KVv*(HDd/2)), 256, 0, stream>>>(xk_f, f_cos, f_sin, k_r, KVv);
  vtrans_k<<<elgrid((LL)Bb*KVv*HDd*Ss), 256, 0, stream>>>(xv_f, v_t);

  // ---- 4) attention: scores = (q @ k^T)/sqrt(HD), softmax, P @ V ----
  gemmB(q_r, HDd, 1, (LL)Ss*HDd,
        k_r, HDd, REP, (LL)Ss*HDd,
        scr_f, Ss, 1, (LL)Ss*Ss, 1, 0,
        nullptr, 0,
        Ss, Ss, HDd, 0.08838834764831845f, Bb*Hh);
  softmax_k<<<dim3((unsigned)(Bb*Hh*Ss)), 256, 0, stream>>>(scr_f, f_mask, attn_b);
  gemmB(attn_b, Ss, 1, (LL)Ss*Ss,
        v_t, Ss, REP, (LL)HDd*Ss,
        ao_f, Dd, Hh, (LL)Ss*Dd, Hh, (LL)HDd,   // scatter into (B,S,H*HD)
        ao_b, Dd,
        Ss, HDd, Ss, 1.0f, Bb*Hh);

  // ---- 5) output projection + LoRA, residual, post-norm ----
  gemm(ao_b, Dd, wo_b, Dd, woo_f, Dd, Tt, Dd, Dd, 1.f, nullptr, 0.f, nullptr, 0);
  gemm(ao_b, Dd, loA_b, Dd, tscr_f, Rr, Tt, Rr, Dd, 1.f, nullptr, 0.f, tq_b, Rr);
  gemm(tq_b, Rr, loB_b, Rr, woo_f, Dd, Tt, Dd, Rr, LSCALE, woo_f, 1.f, nullptr, 0);
  rmsnorm_k<<<Tt, 256, 0, stream>>>(woo_f, f_data, f_ffnw, data2, hn_b, Dd);

  // ---- 6) MoE routing ----
  gemm(hn_b, Dd, gate_b, Dd, lg_f, Ee, Tt, Ee, Dd, 1.f, nullptr, 0.f, nullptr, 0);
  router_k<<<elgrid(Tt), 256, 0, stream>>>(lg_f, rt_f);

  // shared FFN trunks
  gemm(hn_b, Dd, w1_b, Dd, h1_f, FFNf, Tt, FFNf, Dd, 1.f, nullptr, 0.f, nullptr, 0);
  gemm(hn_b, Dd, w3_b, Dd, h3_f, FFNf, Tt, FFNf, Dd, 1.f, nullptr, 0.f, nullptr, 0);

  // ---- 7) expert loop ----
  for (int e = 0; e < Ee; ++e) {
    const bf16_t* eA1 = eA1_b + (LL)e * Rr * Dd;
    const bf16_t* eB1 = eB1_b + (LL)e * FFNf * Rr;
    const bf16_t* eA3 = eA3_b + (LL)e * Rr * Dd;
    const bf16_t* eB3 = eB3_b + (LL)e * FFNf * Rr;
    const bf16_t* eA2 = eA2_b + (LL)e * Rr * FFNf;
    const bf16_t* eB2 = eB2_b + (LL)e * Dd * Rr;

    gemm(hn_b, Dd, eA1, Dd, tscr_f, Rr, Tt, Rr, Dd, 1.f, nullptr, 0.f, tq_b, Rr);
    gemm(tq_b, Rr, eB1, Rr, g1_f, FFNf, Tt, FFNf, Rr, LSCALE, h1_f, 1.f, nullptr, 0);
    gemm(hn_b, Dd, eA3, Dd, tscr_f, Rr, Tt, Rr, Dd, 1.f, nullptr, 0.f, tq_b, Rr);
    gemm(tq_b, Rr, eB3, Rr, g3_f, FFNf, Tt, FFNf, Rr, LSCALE, h3_f, 1.f, nullptr, 0);
    silu_mul_k<<<elgrid((LL)Tt*FFNf), 256, 0, stream>>>(g1_f, g3_f, gh_b, (LL)Tt*FFNf);

    gemm(gh_b, FFNf, w2_b, FFNf, oe_f, Dd, Tt, Dd, FFNf, 1.f, nullptr, 0.f, nullptr, 0);
    gemm(gh_b, FFNf, eA2, FFNf, tscr_f, Rr, Tt, Rr, FFNf, 1.f, nullptr, 0.f, t2_b, Rr);
    gemm(t2_b, Rr, eB2, Rr, oe_f, Dd, Tt, Dd, Rr, LSCALE, oe_f, 1.f, nullptr, 0);

    scale_acc_k<<<elgrid((LL)Tt*Dd), 256, 0, stream>>>(acc_f, oe_f, rt_f, e,
                                                       (e == 0) ? 0.f : 1.f, (LL)Tt*Dd);
  }

  // ---- 8) final residual ----
  add2_k<<<elgrid((LL)Tt*Dd), 256, 0, stream>>>((float*)d_out, data2, acc_f, (LL)Tt*Dd);
}
